// RelativeMultiHeadAttention_44212393345710
// MI455X (gfx1250) — matrix operand order repaired and checked by reading, not compiled, not run on hardware
//
#include <hip/hip_runtime.h>
#include <hip/hip_bf16.h>

typedef __attribute__((ext_vector_type(16))) __bf16 v16bf;
typedef __attribute__((ext_vector_type(8)))  float  v8f;

#define T_SEQ 2048
#define DMODEL 512
#define DHEAD 64
#define NHEAD 8
#define NBATCH 2
#define NROWS (NBATCH * T_SEQ)   // 4096

union ABu { v16bf v; uint4 u[2]; };

// A-operand (16x32 bf16): lane<16 holds K {0..7,16..23}, lane>=16 holds {8..15,24..31}
__device__ inline v16bf load_A16(const __bf16* base, int ldElems, int row, int kbase, int lane) {
    ABu r;
    const __bf16* p = base + (size_t)row * ldElems + kbase + ((lane & 16) ? 8 : 0);
    r.u[0] = *reinterpret_cast<const uint4*>(p);
    r.u[1] = *reinterpret_cast<const uint4*>(p + 16);
    return r.v;
}

// B-operand (32x16 bf16): lane holds one column (n = lane&15),
// lane<16 -> K {0..7,16..23}, lane>=16 -> K {8..15,24..31}
__device__ inline v16bf load_B16(const __bf16* colbase, int kbase, int lane) {
    ABu r;
    const __bf16* p = colbase + kbase + ((lane & 16) ? 8 : 0);
    r.u[0] = *reinterpret_cast<const uint4*>(p);
    r.u[1] = *reinterpret_cast<const uint4*>(p + 16);
    return r.v;
}

__device__ inline v8f wmma_bf16(v16bf a, v16bf b, v8f c) {
    return __builtin_amdgcn_wmma_f32_16x16x32_bf16(false, a, false, b, (short)0, c, false, false);
}

__device__ inline float halfmax16(float x) {
    #pragma unroll
    for (int m = 8; m >= 1; m >>= 1) x = fmaxf(x, __shfl_xor(x, m, 32));
    return x;
}
__device__ inline float halfsum16(float x) {
    #pragma unroll
    for (int m = 8; m >= 1; m >>= 1) x += __shfl_xor(x, m, 32);
    return x;
}

// ---------------- LayerNorm: x [4096,512] -> xn (f32) and xnb (bf16) ----------------
__global__ __launch_bounds__(256) void ln_kernel(const float* __restrict__ x,
                                                 const float* __restrict__ gamma,
                                                 const float* __restrict__ beta,
                                                 float* __restrict__ xn,
                                                 __bf16* __restrict__ xnb) {
    int row = blockIdx.x;
    int tid = threadIdx.x;
    const float* xr = x + (size_t)row * DMODEL;
    float a0 = xr[tid], a1 = xr[tid + 256];
    float s = a0 + a1, sq = a0 * a0 + a1 * a1;
    #pragma unroll
    for (int m = 16; m >= 1; m >>= 1) { s += __shfl_xor(s, m, 32); sq += __shfl_xor(sq, m, 32); }
    __shared__ float ssum[8], ssq[8];
    int wv = tid >> 5, lane = tid & 31;
    if (lane == 0) { ssum[wv] = s; ssq[wv] = sq; }
    __syncthreads();
    if (wv == 0) {
        float ts = (lane < 8) ? ssum[lane] : 0.0f;
        float tq = (lane < 8) ? ssq[lane] : 0.0f;
        #pragma unroll
        for (int m = 4; m >= 1; m >>= 1) { ts += __shfl_xor(ts, m, 32); tq += __shfl_xor(tq, m, 32); }
        if (lane == 0) { ssum[0] = ts; ssq[0] = tq; }
    }
    __syncthreads();
    float mean = ssum[0] * (1.0f / DMODEL);
    float var  = ssq[0] * (1.0f / DMODEL) - mean * mean;
    float rstd = rsqrtf(var + 1e-5f);
    float y0 = (a0 - mean) * rstd * gamma[tid] + beta[tid];
    float y1 = (a1 - mean) * rstd * gamma[tid + 256] + beta[tid + 256];
    size_t o = (size_t)row * DMODEL;
    xn[o + tid] = y0;          xn[o + tid + 256] = y1;
    xnb[o + tid] = (__bf16)y0; xnb[o + tid + 256] = (__bf16)y1;
}

// ---------------- Convert + transpose W [K,N] f32 -> WT [N,K] bf16 ----------------
__global__ void convT_kernel(const float* __restrict__ W, __bf16* __restrict__ WT, int rows, int cols) {
    int idx = blockIdx.x * blockDim.x + threadIdx.x;
    if (idx >= rows * cols) return;
    int r = idx / cols, c = idx % cols;
    WT[(size_t)c * rows + r] = (__bf16)W[idx];
}

// ---------------- Convert lookup table [4095,64] -> bf16 padded to 4096 rows ----------------
__global__ void convE_kernel(const float* __restrict__ lut, __bf16* __restrict__ eb) {
    int idx = blockIdx.x * blockDim.x + threadIdx.x;
    if (idx >= 4096 * DHEAD) return;
    int r = idx >> 6;
    eb[idx] = (r < 2 * T_SEQ - 1) ? (__bf16)lut[idx] : (__bf16)0.0f;
}

// ---------------- Software-pipelined 16x64 GEMM tile body ----------------
// acc[4] += A[16,512] rows m0..m0+15  x  WT columns n0..n0+63 (WT is [N,K] bf16)
__device__ inline void gemm_tile_16x64(const __bf16* __restrict__ Abase, int row,
                                       const __bf16* __restrict__ WT, int n0,
                                       int lane, int n_lo, v8f acc[4]) {
    const __bf16* col0 = WT + (size_t)(n0 + n_lo) * DMODEL;
    const __bf16* col1 = WT + (size_t)(n0 + 16 + n_lo) * DMODEL;
    const __bf16* col2 = WT + (size_t)(n0 + 32 + n_lo) * DMODEL;
    const __bf16* col3 = WT + (size_t)(n0 + 48 + n_lo) * DMODEL;

    v16bf A0 = load_A16(Abase, DMODEL, row, 0, lane);
    v16bf B0 = load_B16(col0, 0, lane);
    v16bf B1 = load_B16(col1, 0, lane);
    v16bf B2 = load_B16(col2, 0, lane);
    v16bf B3 = load_B16(col3, 0, lane);

    #pragma unroll
    for (int kc = 0; kc < 15; ++kc) {
        int kn = (kc + 1) * 32;
        v16bf An  = load_A16(Abase, DMODEL, row, kn, lane);
        v16bf Bn0 = load_B16(col0, kn, lane);
        v16bf Bn1 = load_B16(col1, kn, lane);
        v16bf Bn2 = load_B16(col2, kn, lane);
        v16bf Bn3 = load_B16(col3, kn, lane);
        acc[0] = wmma_bf16(A0, B0, acc[0]);
        acc[1] = wmma_bf16(A0, B1, acc[1]);
        acc[2] = wmma_bf16(A0, B2, acc[2]);
        acc[3] = wmma_bf16(A0, B3, acc[3]);
        A0 = An; B0 = Bn0; B1 = Bn1; B2 = Bn2; B3 = Bn3;
    }
    acc[0] = wmma_bf16(A0, B0, acc[0]);
    acc[1] = wmma_bf16(A0, B1, acc[1]);
    acc[2] = wmma_bf16(A0, B2, acc[2]);
    acc[3] = wmma_bf16(A0, B3, acc[3]);
}

// ---------------- QKV projection: xnb [4096,512] x WT -> q/k [B,H,T,64], vT [B,H,64,T] ----------------
__global__ __launch_bounds__(256) void qkv_kernel(const __bf16* __restrict__ xnb,
                                                  const __bf16* __restrict__ WqT,
                                                  const __bf16* __restrict__ WkT,
                                                  const __bf16* __restrict__ WvT,
                                                  __bf16* __restrict__ qb,
                                                  __bf16* __restrict__ kb,
                                                  __bf16* __restrict__ vTb) {
    int which = blockIdx.y;
    const __bf16* WT = (which == 0) ? WqT : ((which == 1) ? WkT : WvT);
    int lane = threadIdx.x & 31, wave = threadIdx.x >> 5;
    int task = blockIdx.x * 8 + wave;          // 2048 tasks
    int mtile = task >> 3, nquad = task & 7;
    int m0 = mtile * 16, n0 = nquad * 64;
    int n_lo = lane & 15, hlf = lane & 16;
    int row = m0 + n_lo;

    v8f acc[4] = {{}, {}, {}, {}};
    gemm_tile_16x64(xnb, row, WT, n0, lane, n_lo, acc);

    #pragma unroll
    for (int v = 0; v < 8; ++v) {
        int m = v + (hlf ? 8 : 0);
        int tg = m0 + m;
        int b = tg >> 11, t = tg & (T_SEQ - 1);
        #pragma unroll
        for (int nt = 0; nt < 4; ++nt) {
            int colg = n0 + nt * 16 + n_lo;
            int h = colg >> 6, d = colg & 63;
            __bf16 val = (__bf16)acc[nt][v];
            if (which == 2)
                vTb[(size_t)((b * NHEAD + h) * DHEAD + d) * T_SEQ + t] = val;
            else if (which == 0)
                qb[(size_t)((b * NHEAD + h) * T_SEQ + t) * DHEAD + d] = val;
            else
                kb[(size_t)((b * NHEAD + h) * T_SEQ + t) * DHEAD + d] = val;
        }
    }
}

// ---------------- Fused attention (flash-style, one wave per 16-row query tile) ----------------
__global__ __launch_bounds__(256) void attn_kernel(const __bf16* __restrict__ qb,
                                                   const __bf16* __restrict__ kb,
                                                   const __bf16* __restrict__ vTb,
                                                   const __bf16* __restrict__ eb,
                                                   __bf16* __restrict__ ob) {
    __shared__ __align__(16) __bf16 pbuf[8 * 16 * 32];  // per-wave 16x32 P staging
    int lane = threadIdx.x & 31, wave = threadIdx.x >> 5;
    int task = blockIdx.x * 8 + wave;          // 2048 tasks = B*H*128
    int qt = task & 127, bh = task >> 7;
    int t0 = qt * 16;
    const __bf16* Q  = qb  + (size_t)bh * T_SEQ * DHEAD;
    const __bf16* K  = kb  + (size_t)bh * T_SEQ * DHEAD;
    const __bf16* VT = vTb + (size_t)bh * DHEAD * T_SEQ;

    int n_lo = lane & 15, hlf = lane & 16;
    int mrow = t0 + n_lo;
    v16bf Qa0 = load_A16(Q, DHEAD, mrow, 0, lane);
    v16bf Qa1 = load_A16(Q, DHEAD, mrow, 32, lane);

    float rm[8], rs[8], corr[8];
    v8f Oa[4] = {{}, {}, {}, {}};
    #pragma unroll
    for (int v = 0; v < 8; ++v) { rm[v] = -3.0e38f; rs[v] = 0.0f; }

    __bf16* pw = pbuf + wave * 512;
    const float scale = 0.125f;                // 1/sqrt(64)

    for (int s0 = 0; s0 < T_SEQ; s0 += 32) {
        // ---- stage ALL K/E B-tiles first so loads pipeline deeply ----
        const __bf16* kc0 = K + (size_t)(s0 + n_lo) * DHEAD;
        const __bf16* kc1 = K + (size_t)(s0 + 16 + n_lo) * DHEAD;
        int rbase = s0 - t0 + (T_SEQ - 16);    // in [0, 4048]
        const __bf16* e0 = eb + (size_t)(rbase + n_lo) * DHEAD;
        const __bf16* e1 = eb + (size_t)(rbase + 16 + n_lo) * DHEAD;
        const __bf16* e2 = eb + (size_t)(rbase + 32 + n_lo) * DHEAD;

        v16bf Kb00 = load_B16(kc0, 0, lane);
        v16bf Kb01 = load_B16(kc0, 32, lane);
        v16bf Kb10 = load_B16(kc1, 0, lane);
        v16bf Kb11 = load_B16(kc1, 32, lane);
        v16bf Eb00 = load_B16(e0, 0, lane);
        v16bf Eb01 = load_B16(e0, 32, lane);
        v16bf Eb10 = load_B16(e1, 0, lane);
        v16bf Eb11 = load_B16(e1, 32, lane);
        v16bf Eb20 = load_B16(e2, 0, lane);
        v16bf Eb21 = load_B16(e2, 32, lane);

        // ---- content + relative score WMMAs ----
        v8f S0 = {}, S1 = {}, R0 = {}, R1 = {}, R2 = {};
        S0 = wmma_bf16(Qa0, Kb00, S0); S0 = wmma_bf16(Qa1, Kb01, S0);
        S1 = wmma_bf16(Qa0, Kb10, S1); S1 = wmma_bf16(Qa1, Kb11, S1);
        R0 = wmma_bf16(Qa0, Eb00, R0); R0 = wmma_bf16(Qa1, Eb01, R0);
        R1 = wmma_bf16(Qa0, Eb10, R1); R1 = wmma_bf16(Qa1, Eb11, R1);
        R2 = wmma_bf16(Qa0, Eb20, R2); R2 = wmma_bf16(Qa1, Eb21, R2);

        // ---- issue V loads now; they drain behind the softmax VALU work ----
        v16bf Vb0 = load_B16(VT + (size_t)(0  + n_lo) * T_SEQ + s0, 0, lane);
        v16bf Vb1 = load_B16(VT + (size_t)(16 + n_lo) * T_SEQ + s0, 0, lane);
        v16bf Vb2 = load_B16(VT + (size_t)(32 + n_lo) * T_SEQ + s0, 0, lane);
        v16bf Vb3 = load_B16(VT + (size_t)(48 + n_lo) * T_SEQ + s0, 0, lane);

        // ---- skew gather + scale + online softmax ----
        #pragma unroll
        for (int v = 0; v < 8; ++v) {
            int m = v + (hlf ? 8 : 0);
            int jp = 15 + n_lo - m;            // tile0: [0,30]; tile1: jp+16 in [16,46]
            int sl = (jp & 15) | hlf;          // identical source lane for both tiles
            float g0 = __shfl(R0[v], sl, 32);
            float g1 = __shfl(R1[v], sl, 32);
            float g2 = __shfl(R2[v], sl, 32);
            float rel0 = (jp < 16) ? g0 : g1;
            float rel1 = (jp + 16 < 32) ? g1 : g2;
            float s0v = (S0[v] + rel0) * scale;
            float s1v = (S1[v] + rel1) * scale;
            float mx = halfmax16(fmaxf(s0v, s1v));
            float mnew = fmaxf(rm[v], mx);
            float c = __expf(rm[v] - mnew);
            float p0 = __expf(s0v - mnew);
            float p1 = __expf(s1v - mnew);
            rs[v] = rs[v] * c + halfsum16(p0 + p1);
            rm[v] = mnew;
            corr[v] = c;
            pw[m * 32 + n_lo] = (__bf16)p0;
            pw[m * 32 + n_lo + 16] = (__bf16)p1;
        }
        #pragma unroll
        for (int nt = 0; nt < 4; ++nt)
            #pragma unroll
            for (int v = 0; v < 8; ++v) Oa[nt][v] *= corr[v];

        asm volatile("s_wait_dscnt 0" ::: "memory");
        v16bf Pa = load_A16(pw, 32, n_lo, 0, lane);

        // ---- O += P x V ----
        Oa[0] = wmma_bf16(Pa, Vb0, Oa[0]);
        Oa[1] = wmma_bf16(Pa, Vb1, Oa[1]);
        Oa[2] = wmma_bf16(Pa, Vb2, Oa[2]);
        Oa[3] = wmma_bf16(Pa, Vb3, Oa[3]);
    }

    // ---- epilogue: normalize, write bf16 to [B,T,H*64] ----
    int b = bh >> 3, h = bh & 7;
    #pragma unroll
    for (int v = 0; v < 8; ++v) {
        int m = v + (hlf ? 8 : 0);
        float inv = 1.0f / rs[v];
        size_t rowoff = ((size_t)(b * T_SEQ + t0 + m)) * DMODEL + h * DHEAD;
        #pragma unroll
        for (int nt = 0; nt < 4; ++nt)
            ob[rowoff + nt * 16 + n_lo] = (__bf16)(Oa[nt][v] * inv);
    }
}

// ---------------- Output projection + bias + residual: out = xn + O @ Wo + bo ----------------
__global__ __launch_bounds__(256) void proj_kernel(const __bf16* __restrict__ ob,
                                                   const __bf16* __restrict__ WoT,
                                                   const float* __restrict__ bo,
                                                   const float* __restrict__ xn,
                                                   float* __restrict__ out) {
    int lane = threadIdx.x & 31, wave = threadIdx.x >> 5;
    int task = blockIdx.x * 8 + wave;
    int mtile = task >> 3, nquad = task & 7;
    int m0 = mtile * 16, n0 = nquad * 64;
    int n_lo = lane & 15, hlf = lane & 16;
    int row = m0 + n_lo;

    v8f acc[4] = {{}, {}, {}, {}};
    gemm_tile_16x64(ob, row, WoT, n0, lane, n_lo, acc);

    #pragma unroll
    for (int v = 0; v < 8; ++v) {
        int m = v + (hlf ? 8 : 0);
        int tg = m0 + m;
        #pragma unroll
        for (int nt = 0; nt < 4; ++nt) {
            int colg = n0 + nt * 16 + n_lo;
            out[(size_t)tg * DMODEL + colg] = acc[nt][v] + bo[colg] + xn[(size_t)tg * DMODEL + colg];
        }
    }
}

extern "C" void kernel_launch(void* const* d_in, const int* in_sizes, int n_in,
                              void* d_out, int out_size, void* d_ws, size_t ws_size,
                              hipStream_t stream) {
    const float* x     = (const float*)d_in[0];
    const float* lut   = (const float*)d_in[1];
    const float* Wq    = (const float*)d_in[2];
    const float* Wk    = (const float*)d_in[3];
    const float* Wv    = (const float*)d_in[4];
    const float* Wo    = (const float*)d_in[5];
    const float* bo    = (const float*)d_in[6];
    const float* gamma = (const float*)d_in[7];
    const float* beta  = (const float*)d_in[8];
    float* out = (float*)d_out;

    char* ws = (char*)d_ws;
    size_t o = 0;
    float*  xn  = (float*)(ws + o);  o += (size_t)NROWS * DMODEL * 4;           // 8 MB
    __bf16* xnb = (__bf16*)(ws + o); o += (size_t)NROWS * DMODEL * 2;           // 4 MB
    __bf16* WqT = (__bf16*)(ws + o); o += (size_t)DMODEL * DMODEL * 2;
    __bf16* WkT = (__bf16*)(ws + o); o += (size_t)DMODEL * DMODEL * 2;
    __bf16* WvT = (__bf16*)(ws + o); o += (size_t)DMODEL * DMODEL * 2;
    __bf16* WoT = (__bf16*)(ws + o); o += (size_t)DMODEL * DMODEL * 2;
    __bf16* eb  = (__bf16*)(ws + o); o += (size_t)4096 * DHEAD * 2;             // padded
    __bf16* qb  = (__bf16*)(ws + o); o += (size_t)NROWS * DMODEL * 2;
    __bf16* kb  = (__bf16*)(ws + o); o += (size_t)NROWS * DMODEL * 2;
    __bf16* vTb = (__bf16*)(ws + o); o += (size_t)NROWS * DMODEL * 2;
    __bf16* ob  = (__bf16*)(ws + o); o += (size_t)NROWS * DMODEL * 2;

    ln_kernel<<<NROWS, 256, 0, stream>>>(x, gamma, beta, xn, xnb);
    convT_kernel<<<1024, 256, 0, stream>>>(Wq, WqT, DMODEL, DMODEL);
    convT_kernel<<<1024, 256, 0, stream>>>(Wk, WkT, DMODEL, DMODEL);
    convT_kernel<<<1024, 256, 0, stream>>>(Wv, WvT, DMODEL, DMODEL);
    convT_kernel<<<1024, 256, 0, stream>>>(Wo, WoT, DMODEL, DMODEL);
    convE_kernel<<<1024, 256, 0, stream>>>(lut, eb);
    qkv_kernel<<<dim3(256, 3), 256, 0, stream>>>(xnb, WqT, WkT, WvT, qb, kb, vTb);
    attn_kernel<<<256, 256, 0, stream>>>(qb, kb, vTb, eb, ob);
    proj_kernel<<<256, 256, 0, stream>>>(ob, WoT, bo, xn, out);
}
